// metric_52467320488009
// MI455X (gfx1250) — matrix operand order repaired and checked by reading, not compiled, not run on hardware
//
#include <hip/hip_runtime.h>
#include <hip/hip_bf16.h>
#include <float.h>
#include <stdint.h>

// ---------- WMMA types ----------
typedef __attribute__((ext_vector_type(16))) __bf16 v16bf;
typedef __attribute__((ext_vector_type(8)))  float  v8f;

static __device__ inline v8f wmma_bf16(v16bf a, v16bf b, v8f c) {
    // D = A(16x32 bf16) * B(32x16 bf16) + C(16x16 f32)
    return __builtin_amdgcn_wmma_f32_16x16x32_bf16(
        /*neg_a=*/false, a, /*neg_b=*/false, b,
        /*c_mod=*/(short)0, c, /*reuse_a=*/false, /*reuse_b=*/false);
}

// ---------- constants ----------
#define NROWS   16384      // N == M == 16384
#define DDIM    64
#define NTILES  (NROWS / 16)   // 1024 tiles of 16 rows
#define NSPLIT  8              // N-sweep split factor (grid.y)
#define WPB     4              // waves (m-tiles) per block

// bf16 round-to-nearest-even, returns 16-bit pattern
static __device__ inline uint32_t bf16_rne(float f) {
    uint32_t u = __float_as_uint(f);
    return (u + 0x7FFFu + ((u >> 16) & 1u)) >> 16;
}
static __device__ inline float bf16_to_f(uint32_t b) {
    return __uint_as_float(b << 16);
}

// ---------- kernel: init per-row min array to +FLT_MAX ----------
__global__ void init_min_kernel(unsigned int* minarr) {
    int i = blockIdx.x * blockDim.x + threadIdx.x;
    if (i < NROWS) minarr[i] = __float_as_uint(FLT_MAX);
}

// ---------- kernel: squared row norms ----------
__global__ void rownorm_kernel(const float* __restrict__ src, float* __restrict__ dst) {
    int r = blockIdx.x * blockDim.x + threadIdx.x;
    if (r >= NROWS) return;
    const float4* p = (const float4*)(src + r * DDIM);
    float s = 0.f;
#pragma unroll
    for (int i = 0; i < DDIM / 4; ++i) {
        float4 v = p[i];
        s += v.x * v.x + v.y * v.y + v.z * v.z + v.w * v.w;
    }
    dst[r] = s;
}

// ---------- kernel: split fp32 -> bf16 hi/lo, packed in WMMA operand layout ----------
// isA != 0 : A-matrix 16x32 bf16 layout (ISA 7.12.2):
//   lanes 0-15 : M=lane,   V0..3 -> K=0..7,  V4..7 -> K=16..23
//   lanes16-31 : M=lane-16,V0..3 -> K=8..15, V4..7 -> K=24..31
// isA == 0 : B-matrix 32x16 bf16 layout (same K order as A):
//   lanes 0-15 : N=lane,   V0..3 -> K=0..7,  V4..7 -> K=16..23
//   lanes16-31 : N=lane-16,V0..3 -> K=8..15, V4..7 -> K=24..31
// Storage: dword index = ((tile*2 + kc)*32 + lane)*8 + v   (1024 B per tile-chunk)
__global__ void pack_kernel(const float* __restrict__ src,
                            uint32_t* __restrict__ dhi,
                            uint32_t* __restrict__ dlo,
                            int isA) {
    int gid = blockIdx.x * blockDim.x + threadIdx.x;   // NTILES*2*32*8 threads
    if (gid >= NTILES * 2 * 32 * 8) return;
    int v    = gid & 7;
    int lane = (gid >> 3) & 31;
    int kc   = (gid >> 8) & 1;
    int tile = gid >> 9;

    int row = tile * 16 + (lane & 15);
    int grp = lane >> 4;
    int kbase;
    if (isA) kbase = (v < 4) ? (2 * v + 8 * grp) : (16 + 2 * (v - 4) + 8 * grp);
    else     kbase = (v < 4) ? (2 * v + 8 * grp) : (16 + 2 * (v - 4) + 8 * grp);
    int k0 = kc * 32 + kbase;           // even; pair (k0, k0+1) -> [15:0],[31:16]

    float f0 = src[row * DDIM + k0];
    float f1 = src[row * DDIM + k0 + 1];

    uint32_t h0 = bf16_rne(f0), h1 = bf16_rne(f1);
    uint32_t l0 = bf16_rne(f0 - bf16_to_f(h0));
    uint32_t l1 = bf16_rne(f1 - bf16_to_f(h1));

    int idx = ((tile * 2 + kc) * 32 + lane) * 8 + v;
    dhi[idx] = h0 | (h1 << 16);
    dlo[idx] = l0 | (l1 << 16);
}

// ---------- main kernel: WMMA sweep + running min of d^2 ----------
__global__ void igd_main_kernel(const uint32_t* __restrict__ Ahi,
                                const uint32_t* __restrict__ Alo,
                                const uint32_t* __restrict__ Bhi,
                                const uint32_t* __restrict__ Blo,
                                const float* __restrict__ pfsq,
                                const float* __restrict__ xsq,
                                unsigned int* __restrict__ minarr) {
    const int mt   = blockIdx.x * WPB + threadIdx.y;   // m-tile of this wave
    const int lane = threadIdx.x;                      // 0..31 (wave32)
    const int grp  = lane >> 4;

    // operand load: 8 consecutive dwords per lane (32 B -> 2x b128)
    auto ldop = [&](const uint32_t* base, int tile, int kc) -> v16bf {
        return *(const v16bf*)(base + (tile * 2 + kc) * 256 + lane * 8);
    };

    // A (pf) operands live in registers for the whole N sweep
    v16bf a_h0 = ldop(Ahi, mt, 0), a_h1 = ldop(Ahi, mt, 1);
    v16bf a_l0 = ldop(Alo, mt, 0), a_l1 = ldop(Alo, mt, 1);

    // C/D layout: VGPR r, lanes 0-15 -> M=r ; lanes 16-31 -> M=r+8
    float psq[8];
#pragma unroll
    for (int r = 0; r < 8; ++r) psq[r] = pfsq[mt * 16 + r + 8 * grp];

    float dmin[8];
#pragma unroll
    for (int r = 0; r < 8; ++r) dmin[r] = FLT_MAX;

    for (int nt = blockIdx.y; nt < NTILES; nt += NSPLIT) {
        v16bf b_h0 = ldop(Bhi, nt, 0), b_h1 = ldop(Bhi, nt, 1);
        v16bf b_l0 = ldop(Blo, nt, 0), b_l1 = ldop(Blo, nt, 1);
        float xq = xsq[nt * 16 + (lane & 15)];

        v8f c = {};
        // S = Ahi*Bhi + Ahi*Blo + Alo*Bhi  (fp32 accumulate), K = 0..31 then 32..63
        c = wmma_bf16(a_h0, b_h0, c);
        c = wmma_bf16(a_h0, b_l0, c);
        c = wmma_bf16(a_l0, b_h0, c);
        c = wmma_bf16(a_h1, b_h1, c);
        c = wmma_bf16(a_h1, b_l1, c);
        c = wmma_bf16(a_l1, b_h1, c);

#pragma unroll
        for (int r = 0; r < 8; ++r) {
            float d2 = fmaxf(psq[r] + xq - 2.0f * c[r], 0.0f);
            dmin[r] = fminf(dmin[r], d2);
        }
    }

    // min across the 16 lanes of each half (halves hold different M rows)
#pragma unroll
    for (int m = 1; m <= 8; m <<= 1) {
#pragma unroll
        for (int r = 0; r < 8; ++r)
            dmin[r] = fminf(dmin[r], __shfl_xor(dmin[r], m, 32));
    }

    if ((lane & 15) == 0) {
#pragma unroll
        for (int r = 0; r < 8; ++r)
            atomicMin(&minarr[mt * 16 + r + 8 * grp], __float_as_uint(dmin[r]));
        // valid: d^2 >= 0, so uint ordering == float ordering
    }
}

// ---------- finalize: mean of sqrt(min d^2) ----------
__global__ void finalize_kernel(const unsigned int* __restrict__ minarr, float* __restrict__ out) {
    __shared__ float sm[256];
    float acc = 0.f;
    for (int i = threadIdx.x; i < NROWS; i += 256)
        acc += sqrtf(__uint_as_float(minarr[i]));
    sm[threadIdx.x] = acc;
    __syncthreads();
    for (int s = 128; s > 0; s >>= 1) {
        if (threadIdx.x < s) sm[threadIdx.x] += sm[threadIdx.x + s];
        __syncthreads();
    }
    if (threadIdx.x == 0) out[0] = sm[0] / (float)NROWS;
}

extern "C" void kernel_launch(void* const* d_in, const int* in_sizes, int n_in,
                              void* d_out, int out_size, void* d_ws, size_t ws_size,
                              hipStream_t stream) {
    const float* x  = (const float*)d_in[0];   // [16384, 64]
    const float* pf = (const float*)d_in[1];   // [16384, 64]
    float* out = (float*)d_out;

    // workspace layout (bytes)
    char* ws = (char*)d_ws;
    unsigned int* minarr = (unsigned int*)(ws + 0);            // 64 KB
    float* xsq  = (float*)(ws + (64 << 10));                   // 64 KB
    float* pfsq = (float*)(ws + (128 << 10));                  // 64 KB
    size_t packBytes = (size_t)NTILES * 2 * 1024;              // 2 MB each
    uint32_t* Bhi = (uint32_t*)(ws + (192 << 10));
    uint32_t* Blo = (uint32_t*)(ws + (192 << 10) + packBytes);
    uint32_t* Ahi = (uint32_t*)(ws + (192 << 10) + 2 * packBytes);
    uint32_t* Alo = (uint32_t*)(ws + (192 << 10) + 3 * packBytes);

    // 1) reset per-row min (ws is not re-poisoned between replays -> must re-init)
    init_min_kernel<<<(NROWS + 255) / 256, 256, 0, stream>>>(minarr);

    // 2) squared row norms
    rownorm_kernel<<<(NROWS + 255) / 256, 256, 0, stream>>>(x,  xsq);
    rownorm_kernel<<<(NROWS + 255) / 256, 256, 0, stream>>>(pf, pfsq);

    // 3) bf16 hi/lo split + WMMA-operand packing
    int packThreads = NTILES * 2 * 32 * 8;                     // 524288
    pack_kernel<<<(packThreads + 255) / 256, 256, 0, stream>>>(x,  Bhi, Blo, /*isA=*/0);
    pack_kernel<<<(packThreads + 255) / 256, 256, 0, stream>>>(pf, Ahi, Alo, /*isA=*/1);

    // 4) WMMA distance sweep: grid = (m-tiles / WPB, NSPLIT), block = (32, WPB)
    dim3 grid(NTILES / WPB, NSPLIT);
    dim3 block(32, WPB);
    igd_main_kernel<<<grid, block, 0, stream>>>(Ahi, Alo, Bhi, Blo, pfsq, xsq, minarr);

    // 5) mean of min distances
    finalize_kernel<<<1, 256, 0, stream>>>(minarr, out);
}
